// Module_69071664054720
// MI455X (gfx1250) — matrix operand order repaired and checked by reading, not compiled, not run on hardware
//
#include <hip/hip_runtime.h>
#include <hip/hip_bf16.h>

// ---------------- problem constants ----------------
// B=2, T0=1488, T1=48, T=1536, W0=2048, W1=1024, NH=8, KV=1, HD=256
#define TT      1536
#define T0_     1488
#define T1_     48
#define W0_     2048
#define W1_     1024
#define NH_     8
#define HD_     256

typedef __attribute__((ext_vector_type(16))) __bf16 v16bf;
typedef __attribute__((ext_vector_type(8)))  float  v8f;
typedef __attribute__((ext_vector_type(2)))  __bf16 bf16x2;
typedef __attribute__((ext_vector_type(4)))  __bf16 bf16x4;

// =====================================================================
// Generic batched-strided GEMM, C = A[MxK] * op(B)[KxN], f32 in/out,
// v_wmma_f32_16x16x32_bf16 accumulation.  256 threads = 8 wave32.
// Block tile: 16 (M) x 256 (N); each wave owns TWO 16x16 C tiles
// (independent accumulators -> two pipelined WMMAs per K-step sharing
// one A fragment).  K stepped by 32.  LDS tiles are kept in WMMA
// fragment order so each lane reads a v16bf fragment as 2x ds_load_b128.
// Dims: M % 16 == 0, N % 256 == 0, K % 32 == 0 (true for all calls).
// Batch index z = z1*Z0 + z0 with independent stride pairs per operand.
// =====================================================================
template<int TRANSB>
__global__ __launch_bounds__(256)
void gemm_wmma_bf16(const float* __restrict__ A, long long lda, long long sA0, long long sA1,
                    const float* __restrict__ Bm, long long ldb, long long sB0, long long sB1,
                    float* __restrict__ C, long long ldc, long long sC0, long long sC1,
                    int M, int N, int K, int Z0)
{
    const int z  = blockIdx.z;
    const int z0 = z % Z0;
    const int z1 = z / Z0;
    A  += (long long)z0 * sA0 + (long long)z1 * sA1;
    Bm += (long long)z0 * sB0 + (long long)z1 * sB1;
    C  += (long long)z0 * sC0 + (long long)z1 * sC1;

    const int tileM = blockIdx.y * 16;
    const int tileN = blockIdx.x * 256;
    const int tid   = threadIdx.x;
    const int wave  = tid >> 5;
    const int lane  = tid & 31;

    // Fragment-order LDS:
    //   Af[lane][e]        : A 16x32  tile  (1 KB)
    //   Bf[sub][lane][e]   : B 32x256 tile  (16 KB), sub = N/16 subtile 0..15
    __shared__ __align__(32) __bf16 Af[32][16];
    __shared__ __align__(32) __bf16 Bf[16][32][16];
    __bf16* const bfbase = &Bf[0][0][0];

    // ---------------- A staging plan: one float2 per thread ----------------
    // element (ar, ak0), ak0 even.  Fragment dest:
    //   lane = ar + 16*(ak0>=16),  e = ak0&15  (pair e,e+1)
    const int aIdx  = tid * 2;             // 0..510
    const int ar    = aIdx >> 5;           // 0..15
    const int ak0   = aIdx & 31;           // even
    const float* aptr = A + (long long)(tileM + ar) * lda + ak0;
    const int aLane = ar + ((ak0 >= 16) ? 16 : 0);
    const int aE    = ak0 & 15;
    bf16x2* const aDst = (bf16x2*)&Af[aLane][aE];

    // ---------------- B staging plan (32 elements / thread; TRANSB: 16) ----------------
    // Single global base; per-j addresses differ by exact multiples of ldb
    // (foldable immediates).  Single LDS base + small immediate offsets.
    const float* bbase;
    long long bstep;
    int bOff[4];        // !TRANSB: elem offset of (j, u=0); TRANSB: bOff[0] + 1024*j
    if (!TRANSB) {
        // pair-quad qi = tid + j*256 : cc0=(tid&63)*4, kr0 = 2*(tid>>6) + 8*j
        // loads rows kr0, kr0+1 (float4 each), packs K-pairs -> b32 LDS stores
        const int cc0  = (tid & 63) * 4;   // 0..252, constant over j
        const int krp0 = tid >> 6;         // 0..3
        bbase = Bm + (long long)(2 * krp0) * ldb + (tileN + cc0);
        bstep = 32LL * ldb;
#pragma unroll
        for (int j = 0; j < 4; ++j) {
            const int kr0   = 2 * krp0 + 8 * j;
            const int khalf = (kr0 >= 16) ? 16 : 0;
            bOff[j] = ((cc0 >> 4) * 32 + (cc0 & 15) + khalf) * 16 + (kr0 & 15);
        }
    } else {
        // quad qi = tid + j*256 : kr0=(tid&7)*4 (contig in global), cc=(tid>>3)+32*j
        const int kr0 = (tid & 7) * 4;
        const int ccb = tid >> 3;          // 0..31
        bbase = Bm + (long long)(tileN + ccb) * ldb + kr0;
        bstep = 32;
        const int khalf = (kr0 >= 16) ? 16 : 0;
        bOff[0] = ((ccb >> 4) * 32 + (ccb & 15) + khalf) * 16 + (kr0 & 15);
        bOff[1] = bOff[0] + 1024;          // cc += 32 -> subtile group += 2
        bOff[2] = bOff[0] + 2048;
        bOff[3] = bOff[0] + 3072;
    }

    v8f acc0 = {};
    v8f acc1 = {};

    for (int k0 = 0; k0 < K; k0 += 32) {
        // ---- stage A (float2 -> packed bf16 pair) ----
        const float2 a2 = *(const float2*)aptr;
        __builtin_prefetch(aptr + 32, 0, 1);
        aptr += 32;
        {
            bf16x2 p; p[0] = (__bf16)a2.x; p[1] = (__bf16)a2.y;
            *aDst = p;
        }
        // ---- stage B ----
        if (!TRANSB) {
#pragma unroll
            for (int j = 0; j < 4; ++j) {
                const float4 ra = *(const float4*)(bbase + (long long)(8 * j) * ldb);
                const float4 rb = *(const float4*)(bbase + (long long)(8 * j + 1) * ldb);
                const float* raf = (const float*)&ra;
                const float* rbf = (const float*)&rb;
#pragma unroll
                for (int u = 0; u < 4; ++u) {
                    bf16x2 p; p[0] = (__bf16)raf[u]; p[1] = (__bf16)rbf[u];
                    *(bf16x2*)(bfbase + bOff[j] + u * 16) = p;   // K-pair (e,e+1)
                }
            }
        } else {
#pragma unroll
            for (int j = 0; j < 4; ++j) {
                const float4 r = *(const float4*)(bbase + 32LL * j * ldb);
                const float* rf = (const float*)&r;
                bf16x4 p;
                p[0] = (__bf16)rf[0]; p[1] = (__bf16)rf[1];
                p[2] = (__bf16)rf[2]; p[3] = (__bf16)rf[3];
                *(bf16x4*)(bfbase + bOff[j]) = p;                // 8B-aligned b64
            }
        }
        bbase += bstep;

        __syncthreads();

        // ---- fragments: contiguous 32B per lane -> 2x ds_load_b128 each ----
        const v16bf afrag = *(const v16bf*)&Af[lane][0];
        const v16bf bf0   = *(const v16bf*)&Bf[2 * wave][lane][0];
        const v16bf bf1   = *(const v16bf*)&Bf[2 * wave + 1][lane][0];
        acc0 = __builtin_amdgcn_wmma_f32_16x16x32_bf16(
                   false, afrag, false, bf0, (short)0, acc0, false, false);
        acc1 = __builtin_amdgcn_wmma_f32_16x16x32_bf16(
                   false, afrag, false, bf1, (short)0, acc1, false, false);
        __syncthreads();
    }

    // ---- store C: row M = i + 8*(lane>=16), cols n0 and n0+16 ----
    const int n0    = tileN + wave * 32 + (lane & 15);
    const int mbase = tileM + ((lane >> 4) << 3);
#pragma unroll
    for (int i = 0; i < 8; ++i) {
        float* crow = C + (long long)(mbase + i) * ldc + n0;
        crow[0]  = acc0[i];
        crow[16] = acc1[i];
    }
}

// =====================================================================
// RoPE: 128 threads per (b, t, head); pair (i, i+128) rotated
// =====================================================================
__global__ __launch_bounds__(128)
void rope_kernel(const float* __restrict__ src, float* __restrict__ dst,
                 const int* __restrict__ positions, int T, int heads, float scale)
{
    const int idx = blockIdx.x;
    const int n   = idx % heads;
    const int bt  = idx / heads;
    const int t   = bt % T;
    const int b   = bt / T;
    const int i   = threadIdx.x;   // 0..127

    const float pos = (float)positions[b * T + t];
    const float fe  = (2.0f / 256.0f) * (float)i;
    const float r   = pos * __expf(fe * -9.2103403719761836f); // 1/10000^fe
    const float sn  = __sinf(r);
    const float cs  = __cosf(r);

    const long long base = ((long long)(b * T + t) * heads + n) * 256;
    const float x1 = src[base + i];
    const float x2 = src[base + i + 128];
    dst[base + i]       = (x1 * cs - x2 * sn) * scale;
    dst[base + i + 128] = (x2 * cs + x1 * sn) * scale;
}

// =====================================================================
// Causal softmax, in place on the probs region. One block per row.
// Masked positions (s > t) written as exact 0 (= softmax of BIG_NEG).
// =====================================================================
__global__ __launch_bounds__(256)
void softmax_causal(float* __restrict__ probs, int T)
{
    const long long row = blockIdx.x;          // (b*NH + h)*T + t
    const int t   = (int)(row % T);
    float* p      = probs + row * (long long)T;
    const int tid = threadIdx.x;

    __shared__ float red[256];

    float vmax = -3.0e38f;
    for (int s = tid; s <= t; s += 256) vmax = fmaxf(vmax, p[s]);
    red[tid] = vmax;
    __syncthreads();
    for (int off = 128; off > 0; off >>= 1) {
        if (tid < off) red[tid] = fmaxf(red[tid], red[tid + off]);
        __syncthreads();
    }
    vmax = red[0];
    __syncthreads();

    float sum = 0.0f;
    for (int s = tid; s <= t; s += 256) sum += __expf(p[s] - vmax);
    red[tid] = sum;
    __syncthreads();
    for (int off = 128; off > 0; off >>= 1) {
        if (tid < off) red[tid] += red[tid + off];
        __syncthreads();
    }
    const float inv = 1.0f / red[0];
    __syncthreads();

    for (int s = tid; s < T; s += 256)
        p[s] = (s <= t) ? __expf(p[s] - vmax) * inv : 0.0f;
}

// =====================================================================
// host-side launcher
// =====================================================================
static inline void launch_gemm(const float* A, long long lda, long long sA0, long long sA1,
                               const float* B, long long ldb, long long sB0, long long sB1, int transB,
                               float* C, long long ldc, long long sC0, long long sC1,
                               int M, int N, int K, int Z0, int Z, hipStream_t s)
{
    dim3 grid(N / 256, M / 16, Z);
    if (transB)
        gemm_wmma_bf16<1><<<grid, 256, 0, s>>>(A, lda, sA0, sA1, B, ldb, sB0, sB1,
                                               C, ldc, sC0, sC1, M, N, K, Z0);
    else
        gemm_wmma_bf16<0><<<grid, 256, 0, s>>>(A, lda, sA0, sA1, B, ldb, sB0, sB1,
                                               C, ldc, sC0, sC1, M, N, K, Z0);
}

extern "C" void kernel_launch(void* const* d_in, const int* in_sizes, int n_in,
                              void* d_out, int out_size, void* d_ws, size_t ws_size,
                              hipStream_t stream)
{
    (void)in_sizes; (void)n_in; (void)out_size; (void)ws_size;

    const float* x0        = (const float*)d_in[0];
    const float* x1        = (const float*)d_in[1];
    const int*   positions = (const int*)d_in[2];
    /* d_in[3] attn_mask: causal by construction, not needed */
    const float* wq0  = (const float*)d_in[4];
    const float* wkv0 = (const float*)d_in[5];
    const float* wo0  = (const float*)d_in[6];
    const float* wq1  = (const float*)d_in[7];
    const float* wkv1 = (const float*)d_in[8];
    const float* wo1  = (const float*)d_in[9];

    float* out = (float*)d_out;
    float* ws  = (float*)d_ws;

    // ---- workspace layout (floats) ----
    float* q    = ws;                    // (B,T,8,256)   6,291,456
    float* kraw = ws + 6291456LL;        // (B,T,256)       786,432
    float* enc  = ws + 7077888LL;        // (B,T,8,256)   6,291,456

    // ---- d_out layout (floats, return-order concat) ----
    float* out0  = out;                  // (2,1488,2048)
    float* out1  = out + 6094848LL;      // (2,48,1024)
    float* outk  = out + 6193152LL;      // (2,1536,1,256)
    float* outv  = out + 6979584LL;      // (2,1536,1,256)
    float* probs = out + 7766016LL;      // (2,1,8,1536,1536)
    float* hoe   = out + 45514752LL;     // (2,48,8,2048)

    const long long sX0 = (long long)T0_ * W0_;       // 3047424
    const long long sX1 = (long long)T1_ * W1_;       // 49152
    const long long sQ  = (long long)TT * NH_ * HD_;  // 3145728
    const long long sKV = (long long)TT * HD_;        // 393216
    const long long sP  = (long long)TT * TT;         // 2359296

    // 1) q0 = x0 @ wq0  -> q rows [0,1488)         per (b, head)
    launch_gemm(x0, W0_, 0, sX0,
                wq0, HD_, (long long)W0_ * HD_, 0, 0,
                q, NH_ * HD_, HD_, sQ,
                T0_, HD_, W0_, NH_, 2 * NH_, stream);
    // 2) q1 = x1 @ wq1  -> q rows [1488,1536)
    launch_gemm(x1, W1_, 0, sX1,
                wq1, HD_, (long long)W1_ * HD_, 0, 0,
                q + (long long)T0_ * NH_ * HD_, NH_ * HD_, HD_, sQ,
                T1_, HD_, W1_, NH_, 2 * NH_, stream);
    // 3) k0 = x0 @ wkv0[0]  -> kraw (pre-RoPE)     per b
    launch_gemm(x0, W0_, 0, sX0,
                wkv0, HD_, 0, 0, 0,
                kraw, HD_, 0, sKV,
                T0_, HD_, W0_, 1, 2, stream);
    // 4) v0 = x0 @ wkv0[1]  -> d_out v region
    launch_gemm(x0, W0_, 0, sX0,
                wkv0 + (long long)W0_ * HD_, HD_, 0, 0, 0,
                outv, HD_, 0, sKV,
                T0_, HD_, W0_, 1, 2, stream);
    // 5) k1 = x1 @ wkv1[0]
    launch_gemm(x1, W1_, 0, sX1,
                wkv1, HD_, 0, 0, 0,
                kraw + (long long)T0_ * HD_, HD_, 0, sKV,
                T1_, HD_, W1_, 1, 2, stream);
    // 6) v1 = x1 @ wkv1[1]
    launch_gemm(x1, W1_, 0, sX1,
                wkv1 + (long long)W1_ * HD_, HD_, 0, 0, 0,
                outv + (long long)T0_ * HD_, HD_, 0, sKV,
                T1_, HD_, W1_, 1, 2, stream);

    // 7) RoPE on q (in place, * HEAD_DIM^-0.5 = 1/16)
    rope_kernel<<<2 * TT * NH_, 128, 0, stream>>>(q, q, positions, TT, NH_, 0.0625f);
    // 8) RoPE on k: kraw -> d_out k region
    rope_kernel<<<2 * TT, 128, 0, stream>>>(kraw, outk, positions, TT, 1, 1.0f);

    // 9) logits = q @ k^T  -> probs region (raw)   per (b, head), transB
    launch_gemm(q, NH_ * HD_, HD_, sQ,
                outk, HD_, 0, sKV, 1,
                probs, TT, sP, NH_ * sP,
                TT, TT, HD_, NH_, 2 * NH_, stream);

    // 10) causal softmax in place -> final probs
    softmax_causal<<<2 * NH_ * TT, 256, 0, stream>>>(probs, TT);

    // 11) encoded = probs @ v  -> enc              per (b, head)
    launch_gemm(probs, TT, sP, NH_ * sP,
                outv, HD_, 0, sKV, 0,
                enc, NH_ * HD_, HD_, sQ,
                TT, HD_, TT, NH_, 2 * NH_, stream);

    // 12) out0 = sum_n enc[:, :1488] @ wo0  (single K=2048 GEMM, wo0 flat) per b
    launch_gemm(enc, NH_ * HD_, 0, sQ,
                wo0, W0_, 0, 0, 0,
                out0, W0_, 0, (long long)T0_ * W0_,
                T0_, W0_, NH_ * HD_, 1, 2, stream);
    // 13) out1 = sum_n enc[:, 1488:] @ wo1  per b
    launch_gemm(enc + (long long)T0_ * NH_ * HD_, NH_ * HD_, 0, sQ,
                wo1, W1_, 0, 0, 0,
                out1, W1_, 0, (long long)T1_ * W1_,
                T1_, W1_, NH_ * HD_, 1, 2, stream);
    // 14) head_output_encoder = per-head enc[:,1440:1488] @ wo0[n]  per (b, head)
    launch_gemm(enc + 1440LL * NH_ * HD_, NH_ * HD_, HD_, sQ,
                wo0, W0_, (long long)HD_ * W0_, 0, 0,
                hoe, (long long)NH_ * W0_, W0_, (long long)T1_ * NH_ * W0_,
                T1_, W0_, HD_, NH_, 2 * NH_, stream);
}
